// Capsule_20555713479182
// MI455X (gfx1250) — matrix operand order repaired and checked by reading, not compiled, not run on hardware
//
#include <hip/hip_runtime.h>
#include <hip/hip_bf16.h>

// ---------------------------------------------------------------------------
// Capsule attention module, reduced:
//   softmax over a size-1 axis == 1.0, so all ROUTINGS iterations collapse to
//     s[b,d] = sum_{j,e} W[0,j,d,e] * x[b,j,e]   (GEMM M=512,N=1024,K=36864)
//     out    = squash(s)  over d
// fp32->f16 convert into ws, f16 WMMA GEMM (v_wmma_f32_16x16x32_f16, f32
// accumulate), split-K=8 partial slabs, fused reduce+squash epilogue.
// Converted operands (113MB) are L2-resident (192MB L2).
// GEMM main loop: double-buffered LDS, BK=128, and global loads are issued
// AFTER each barrier so the backend's conservative wait-before-barrier only
// sees dscnt; the loadcnt wait lands after the 16-WMMA compute block.
// ---------------------------------------------------------------------------

typedef _Float16 h16 __attribute__((ext_vector_type(16)));
typedef _Float16 h8  __attribute__((ext_vector_type(8)));
typedef _Float16 h4  __attribute__((ext_vector_type(4)));
typedef float    f8  __attribute__((ext_vector_type(8)));

#define BATCH_SZ 512
#define NCAPS    36
#define DDIM     1024
#define KTOT     (NCAPS * DDIM)       // 36864
#define SPLITK   8
#define KC       (KTOT / SPLITK)      // 4608; KC/BK = 36 -> 18 exact double-stages
#define BM       128
#define BN       64
#define BK       128
#define LDA      (BK + 8)             // 136 halves: 68-dword row stride (==4 mod 8)
#define LDB      (BK + 8)
#define NT       (DDIM / BN)          // 16 N-tiles
#define MT       (BATCH_SZ / BM)      // 4  M-tiles

// ---------------- fp32 -> f16 conversion (vectorized x4) -------------------
__global__ __launch_bounds__(256) void cvt_f32_f16(const float* __restrict__ in,
                                                   _Float16* __restrict__ out,
                                                   int n4) {
  int i = blockIdx.x * blockDim.x + threadIdx.x;
  if (i >= n4) return;
  float4 v = ((const float4*)in)[i];
  h4 o;
  o[0] = (_Float16)v.x; o[1] = (_Float16)v.y;
  o[2] = (_Float16)v.z; o[3] = (_Float16)v.w;
  ((h4*)out)[i] = o;
}

// ---------------- GEMM helpers ---------------------------------------------
// Copy assignment: seg = t&15 (16B segment of a 128-half row), rowc = t>>4.
// A tile: 128 rows -> rows rowc + u*16, u=0..7.  B tile: 64 rows, u=0..3.
__device__ __forceinline__ void load_tile(const _Float16* __restrict__ Xh,
                                          const _Float16* __restrict__ Wh,
                                          int m0, int n0, int rowc, int seg, int kg,
                                          h8 a[8], h8 b[4]) {
#pragma unroll
  for (int u = 0; u < 8; ++u)
    a[u] = *(const h8*)(Xh + (size_t)(m0 + rowc + u * 16) * KTOT + kg + seg * 8);
  const int kB = kg + seg * 8;               // BK=128 chunks never cross a j boundary
  const size_t wbase = (size_t)(kB >> 10) * (DDIM * DDIM) + (kB & 1023);
#pragma unroll
  for (int u = 0; u < 4; ++u)
    b[u] = *(const h8*)(Wh + wbase + (size_t)(n0 + rowc + u * 16) * DDIM);
}

__device__ __forceinline__ void store_tile(_Float16* __restrict__ Ab,
                                           _Float16* __restrict__ Bb,
                                           int rowc, int seg,
                                           const h8 a[8], const h8 b[4]) {
#pragma unroll
  for (int u = 0; u < 8; ++u)
    *(h8*)(Ab + (rowc + u * 16) * LDA + seg * 8) = a[u];
#pragma unroll
  for (int u = 0; u < 4; ++u)
    *(h8*)(Bb + (rowc + u * 16) * LDB + seg * 8) = b[u];
}

__device__ __forceinline__ void prefetch_tile(const _Float16* __restrict__ Xh,
                                              const _Float16* __restrict__ Wh,
                                              int m0, int n0, int rowc, int seg, int kg) {
  __builtin_prefetch(Xh + (size_t)(m0 + rowc * 8) * KTOT + kg + seg * 8, 0, 1);
  const int kB = kg + seg * 8;
  __builtin_prefetch(Wh + (size_t)(kB >> 10) * (DDIM * DDIM) +
                     (size_t)(n0 + rowc * 4) * DDIM + (kB & 1023), 0, 1);
}

__device__ __forceinline__ void compute_tile(const _Float16* __restrict__ Ab,
                                             const _Float16* __restrict__ Bb,
                                             int wm, int wn, int r, int part,
                                             f8 acc[2][2]) {
#pragma unroll
  for (int ks = 0; ks < 4; ++ks) {           // BK=128 -> four wmma K-steps of 32
    const int k0 = ks * 32;
    h16 af[2], bf[2];
#pragma unroll
    for (int i = 0; i < 2; ++i) {
      // A fragment: lane (r,part) holds M=r, K = {p*8..p*8+7, 16+p*8..16+p*8+7}
      const _Float16* ap = Ab + (wm + i * 16 + r) * LDA + k0 + part * 8;
      h8 lo = *(const h8*)ap;
      h8 hi = *(const h8*)(ap + 16);
      af[i] = __builtin_shufflevector(lo, hi, 0, 1, 2, 3, 4, 5, 6, 7,
                                      8, 9, 10, 11, 12, 13, 14, 15);
    }
#pragma unroll
    for (int jn = 0; jn < 2; ++jn) {
      // B fragment: lane (r,part) holds column N=r, K = {p*8..p*8+7, 16+p*8..16+p*8+7}
      const _Float16* bp = Bb + (wn + jn * 16 + r) * LDB + k0 + part * 8;
      h8 lo = *(const h8*)bp;
      h8 hi = *(const h8*)(bp + 16);
      bf[jn] = __builtin_shufflevector(lo, hi, 0, 1, 2, 3, 4, 5, 6, 7,
                                       8, 9, 10, 11, 12, 13, 14, 15);
    }
#pragma unroll
    for (int i = 0; i < 2; ++i)
#pragma unroll
      for (int jn = 0; jn < 2; ++jn)
        acc[i][jn] = __builtin_amdgcn_wmma_f32_16x16x32_f16(
            false, af[i], false, bf[jn], (short)0, acc[i][jn], false, false);
  }
}

// ---------------- split-K WMMA GEMM (double-buffered, pipelined) -----------
// s_part[chunk][m][n] = sum_{k in chunk} Xh[m,k] * Wh[j(k), n, e(k)]
__global__ __launch_bounds__(256) void gemm_wmma_f16(
    const _Float16* __restrict__ Xh,   // (512, 36864) row-major
    const _Float16* __restrict__ Wh,   // (36, 1024, 1024) = [j][d][e]
    float* __restrict__ s_part) {      // (SPLITK, 512, 1024)

  __shared__ __align__(16) _Float16 Alds[2][BM * LDA];   // 2 x 34 KB
  __shared__ __align__(16) _Float16 Blds[2][BN * LDB];   // 2 x 17 KB

  const int tile  = blockIdx.x;          // 0..MT*NT-1
  const int chunk = blockIdx.y;          // 0..SPLITK-1
  const int n0 = (tile % NT) * BN;
  const int m0 = (tile / NT) * BM;
  const int kbase = chunk * KC;

  const int t    = threadIdx.x;
  const int lane = t & 31;
  const int wave = t >> 5;               // 8 waves
  const int wm   = (wave & 3) * 32;      // 4 waves along M
  const int wn   = (wave >> 2) * 32;     // 2 waves along N

  const int seg  = t & 15;               // 8-half (16B) segment within a BK row
  const int rowc = t >> 4;               // 0..15

  const int r    = lane & 15;
  const int part = lane >> 4;

  f8 acc[2][2] = {};
  h8 a[8], b[4];

  // prologue: stage tile 0 into buffer 0
  load_tile(Xh, Wh, m0, n0, rowc, seg, kbase, a, b);
  store_tile(Alds[0], Blds[0], rowc, seg, a, b);

#pragma unroll 1
  for (int kk = 0; kk < KC; kk += 2 * BK) {   // 18 double-stages
    // ---- stage 0: compute buf0, prefill buf1 with tile (kk+BK) ----
    __syncthreads();                          // only dscnt pending -> cheap
    load_tile(Xh, Wh, m0, n0, rowc, seg, kbase + kk + BK, a, b);  // after barrier
    if (kk + 2 * BK < KC)
      prefetch_tile(Xh, Wh, m0, n0, rowc, seg, kbase + kk + 2 * BK);
    compute_tile(Alds[0], Blds[0], wm, wn, r, part, acc);         // hides load
    store_tile(Alds[1], Blds[1], rowc, seg, a, b);                // waits loadcnt here

    // ---- stage 1: compute buf1, prefill buf0 with tile (kk+2*BK) ----
    __syncthreads();
    const bool hn = (kk + 2 * BK) < KC;
    if (hn)
      load_tile(Xh, Wh, m0, n0, rowc, seg, kbase + kk + 2 * BK, a, b);
    if (kk + 3 * BK < KC)
      prefetch_tile(Xh, Wh, m0, n0, rowc, seg, kbase + kk + 3 * BK);
    compute_tile(Alds[1], Blds[1], wm, wn, r, part, acc);
    if (hn)
      store_tile(Alds[0], Blds[0], rowc, seg, a, b);
  }

  // ---- epilogue: C/D layout row = v + 8*part, col = r ----
  float* sp = s_part + (size_t)chunk * BATCH_SZ * DDIM;
#pragma unroll
  for (int i = 0; i < 2; ++i)
#pragma unroll
    for (int jn = 0; jn < 2; ++jn) {
      const int col = n0 + wn + jn * 16 + r;
#pragma unroll
      for (int v = 0; v < 8; ++v) {
        const int row = m0 + wm + i * 16 + part * 8 + v;
        sp[(size_t)row * DDIM + col] = acc[i][jn][v];
      }
    }
}

// ---------------- reduce split-K partials + squash -------------------------
__global__ __launch_bounds__(256) void reduce_squash(const float* __restrict__ s_part,
                                                     float* __restrict__ out) {
  const int b = blockIdx.x;
  const int t = threadIdx.x;
  float v[DDIM / 256];
  float ss = 0.f;
#pragma unroll
  for (int q = 0; q < DDIM / 256; ++q) {
    const int d = t + q * 256;
    float s = 0.f;
#pragma unroll
    for (int c = 0; c < SPLITK; ++c)
      s += s_part[((size_t)c * BATCH_SZ + b) * DDIM + d];
    v[q] = s;
    ss += s * s;
  }
  __shared__ float red[256];
  red[t] = ss;
  __syncthreads();
  for (int off = 128; off > 0; off >>= 1) {
    if (t < off) red[t] += red[t + off];
    __syncthreads();
  }
  const float n2 = red[0];
  const float nrm = sqrtf(n2);
  const float scale = n2 / (1.f + n2) / (nrm + 1e-8f);
#pragma unroll
  for (int q = 0; q < DDIM / 256; ++q)
    out[(size_t)b * DDIM + t + q * 256] = scale * v[q];
}

// ---------------------------------------------------------------------------
extern "C" void kernel_launch(void* const* d_in, const int* in_sizes, int n_in,
                              void* d_out, int out_size, void* d_ws, size_t ws_size,
                              hipStream_t stream) {
  const float* x = (const float*)d_in[0];   // (512, 36, 1024) fp32
  const float* w = (const float*)d_in[1];   // (1, 36, 1024, 1024) fp32
  float* out = (float*)d_out;               // (512, 1, 1024) fp32

  // workspace layout (~124 MB): Wh f16 | Xh f16 | s_part f32
  _Float16* Wh = (_Float16*)d_ws;
  _Float16* Xh = Wh + (size_t)NCAPS * DDIM * DDIM;
  float* s_part = (float*)(Xh + (size_t)BATCH_SZ * KTOT);

  const int nw4 = (NCAPS * DDIM * DDIM) / 4;   // 9,437,184
  const int nx4 = (BATCH_SZ * KTOT) / 4;       // 4,718,592
  cvt_f32_f16<<<(nw4 + 255) / 256, 256, 0, stream>>>(w, Wh, nw4);
  cvt_f32_f16<<<(nx4 + 255) / 256, 256, 0, stream>>>(x, Xh, nx4);

  dim3 grid(MT * NT, SPLITK);                  // (64, 8) = 512 workgroups
  gemm_wmma_f16<<<grid, 256, 0, stream>>>(Xh, Wh, s_part);

  reduce_squash<<<BATCH_SZ, 256, 0, stream>>>(s_part, out);
}
